// Attention_45698452029589
// MI455X (gfx1250) — matrix operand order repaired and checked by reading, not compiled, not run on hardware
//
#include <hip/hip_runtime.h>

#define DIMC   768
#define NHEADS 12
#define HDIM   64
#define BATCH  16
#define SEQ    1024
#define MTOT   (BATCH*SEQ)     // 16384
#define QKVN   (3*DIMC)        // 2304

typedef __attribute__((ext_vector_type(16))) __bf16 v16bf;
typedef __attribute__((ext_vector_type(8)))  float  v8f;
typedef __attribute__((ext_vector_type(4)))  int    v4i;
typedef unsigned short u16;
typedef unsigned int   u32;
typedef unsigned long long u64;

union Frag { v16bf v; uint4 q[2]; };
union V32  { uint4 q[4]; u16 h[32]; };

// ---- CDNA5 async global->LDS path (ASYNCcnt-tracked), with safe fallback ----
#if defined(__has_builtin)
# if __has_builtin(__builtin_amdgcn_global_load_async_to_lds_b128) && \
     __has_builtin(__builtin_amdgcn_s_wait_asynccnt)
#  define HAVE_ASYNC_LDS 1
# endif
#endif
#ifndef HAVE_ASYNC_LDS
# define HAVE_ASYNC_LDS 0
#endif

#define AS1 __attribute__((address_space(1)))
#define AS3 __attribute__((address_space(3)))

__device__ __forceinline__ void cp_b128(u16* ldst, const u16* gsrc) {
#if HAVE_ASYNC_LDS
    __builtin_amdgcn_global_load_async_to_lds_b128(
        (AS1 v4i*)(u64)(const void*)gsrc,
        (AS3 v4i*)(u32)(u64)(const void*)ldst, 0, 0);
#else
    *(uint4*)ldst = *(const uint4*)gsrc;
#endif
}
__device__ __forceinline__ void cp_wait() {
#if HAVE_ASYNC_LDS
    __builtin_amdgcn_s_wait_asynccnt(0);   // my async LDS writes are visible
#endif
}

__device__ __forceinline__ u16 f2bf(float f) {
    u32 u = __builtin_bit_cast(u32, f);
    u32 r = u + 0x7FFFu + ((u >> 16) & 1u);   // round-to-nearest-even
    return (u16)(r >> 16);
}

__device__ __forceinline__ v8f wmma_bf16(const Frag& a, const Frag& b, v8f c) {
    return __builtin_amdgcn_wmma_f32_16x16x32_bf16(
        false, a.v, false, b.v, (short)0, c, false, false);
}

// ---------------- Pass 0: f32 -> bf16 convert (vectorized x4) ----------------
__global__ void cvt_f32_bf16(const float* __restrict__ src, u16* __restrict__ dst, int n4) {
    int i = blockIdx.x * 256 + threadIdx.x;
    if (i < n4) {
        float4 f = reinterpret_cast<const float4*>(src)[i];
        u32 lo = (u32)f2bf(f.x) | ((u32)f2bf(f.y) << 16);
        u32 hi = (u32)f2bf(f.z) | ((u32)f2bf(f.w) << 16);
        reinterpret_cast<uint2*>(dst)[i] = make_uint2(lo, hi);
    }
}

// ---------------- Passes 1 & 3: bf16 WMMA GEMM, C = A * B^T + bias ----------------
// Double-buffered LDS, async global->LDS staging, one barrier per k-step.
// mode 0: scatter bf16 q/k/v   mode 1: write f32 to outf
__global__ __launch_bounds__(256)
void gemm_bf16(const u16* __restrict__ A, const u16* __restrict__ Bmat,
               const float* __restrict__ bias,
               u16* __restrict__ qb, u16* __restrict__ kb, u16* __restrict__ vb,
               float* __restrict__ outf, int K, int mode)
{
    __shared__ __align__(16) u16 Al[2][128 * 32];
    __shared__ __align__(16) u16 Bl[2][128 * 32];

    const int mtile = blockIdx.y, ntile = blockIdx.x;
    const int tid  = threadIdx.x;
    const int wave = tid >> 5, lane = tid & 31;
    const int g = lane >> 4, r = lane & 15;
    const int wm = wave & 3, wn = wave >> 2;        // 4x2 wave grid, 32x64 per wave

    v8f acc[2][4];
    for (int i = 0; i < 2; i++)
        for (int j = 0; j < 4; j++) { v8f z = {}; acc[i][j] = z; }

    const int ldr = tid >> 1;           // 0..127
    const int ldc = (tid & 1) * 16;     // 0 or 16
    const int lo  = ldr * 32 + ldc;

    const u16* Ag = A    + (size_t)(mtile * 128 + ldr) * K + ldc;
    const u16* Bg = Bmat + (size_t)(ntile * 128 + ldr) * K + ldc;

    // prologue: stage k-step 0 into buffer 0
    cp_b128(&Al[0][lo],     Ag);
    cp_b128(&Al[0][lo + 8], Ag + 8);
    cp_b128(&Bl[0][lo],     Bg);
    cp_b128(&Bl[0][lo + 8], Bg + 8);

    const int nIter = K / 32;
    for (int it = 0; it < nIter; it++) {
        const int cb = it & 1, nb = (it + 1) & 1;
        cp_wait();              // my async writes into buffer cb done
        __syncthreads();        // => all waves done writing cb AND done reading nb (it-1 compute)
        if (it + 1 < nIter) {   // prefetch next k-step while computing this one
            int kt = (it + 1) * 32;
            cp_b128(&Al[nb][lo],     Ag + kt);
            cp_b128(&Al[nb][lo + 8], Ag + kt + 8);
            cp_b128(&Bl[nb][lo],     Bg + kt);
            cp_b128(&Bl[nb][lo + 8], Bg + kt + 8);
        }

        Frag af[2], bf[4];
        for (int mi = 0; mi < 2; mi++) {            // A frag: row per lane, interleaved K
            const u16* p = &Al[cb][(wm * 32 + mi * 16 + r) * 32 + g * 8];
            af[mi].q[0] = *(const uint4*)p;
            af[mi].q[1] = *(const uint4*)(p + 16);
        }
        for (int nj = 0; nj < 4; nj++) {            // B frag: col per lane, interleaved K
            const u16* p = &Bl[cb][(wn * 64 + nj * 16 + r) * 32 + g * 8];
            bf[nj].q[0] = *(const uint4*)p;
            bf[nj].q[1] = *(const uint4*)(p + 16);
        }
        for (int mi = 0; mi < 2; mi++)
            for (int nj = 0; nj < 4; nj++)
                acc[mi][nj] = wmma_bf16(af[mi], bf[nj], acc[mi][nj]);
    }

    for (int mi = 0; mi < 2; mi++)
        for (int nj = 0; nj < 4; nj++)
            for (int v = 0; v < 8; v++) {
                int gm = mtile * 128 + wm * 32 + mi * 16 + v + g * 8;
                int gn = ntile * 128 + wn * 64 + nj * 16 + r;
                float val = acc[mi][nj][v] + bias[gn];
                if (mode == 0) {
                    int which = gn / DIMC;
                    int d = gn % DIMC;
                    int h = d / HDIM, hd = d % HDIM;
                    int b = gm / SEQ, tok = gm % SEQ;
                    size_t off = ((size_t)(b * NHEADS + h) * SEQ + tok) * HDIM + hd;
                    u16 bv = f2bf(val);
                    if (which == 0)      qb[off] = bv;
                    else if (which == 1) kb[off] = bv;
                    else                 vb[off] = bv;
                } else {
                    outf[(size_t)gm * DIMC + gn] = val;
                }
            }
}

// ---------------- Pass 2: flash attention, one (b,h) x 128-row q-tile per block ----------------
__global__ __launch_bounds__(256)
void flash_attn(const u16* __restrict__ qb, const u16* __restrict__ kb,
                const u16* __restrict__ vb, u16* __restrict__ attnb)
{
    __shared__ __align__(16) u16 Kl[128 * 64];        // [token][d]
    __shared__ __align__(16) u16 Vt[64 * 128];        // [d][token]  (transposed)
    __shared__ __align__(16) u16 Pl[8 * 16 * 128];    // per-wave P staging

    const int qt  = blockIdx.x;        // 0..7 q tile
    const int bh  = blockIdx.y;        // 0..191
    const int tid = threadIdx.x;
    const int wave = tid >> 5, lane = tid & 31;
    const int g = lane >> 4, r = lane & 15;

    const float cscale = 0.125f * 1.44269504088896340736f;  // 1/sqrt(64) * log2(e)

    // per-wave 16 query rows -> A fragments, kept in registers for the whole loop
    const int qrow = qt * 128 + wave * 16 + r;
    const u16* Qg = qb + ((size_t)bh * SEQ + qrow) * HDIM;
    Frag aq[2];
    for (int kk = 0; kk < 2; kk++) {
        int d0 = kk * 32 + g * 8;
        aq[kk].q[0] = *(const uint4*)(Qg + d0);
        aq[kk].q[1] = *(const uint4*)(Qg + d0 + 16);
    }

    v8f o[4];
    for (int dj = 0; dj < 4; dj++) { v8f z = {}; o[dj] = z; }
    float mi_[8], li_[8];
    for (int v = 0; v < 8; v++) { mi_[v] = -1.0e30f; li_[v] = 0.f; }

    u16* Pw = &Pl[wave * 16 * 128];

    const int ldr = tid >> 1;           // token row 0..127
    const int ldh = (tid & 1) * 32;     // d offset 0 or 32

    for (int kt = 0; kt < SEQ; kt += 128) {
        __syncthreads();                // all waves done reading Kl/Vt of prev iter
        // K tile: async global->LDS (no data VGPRs)
        const u16* Kg = kb + ((size_t)bh * SEQ + kt + ldr) * HDIM + ldh;
        cp_b128(&Kl[ldr * 64 + ldh],      Kg);
        cp_b128(&Kl[ldr * 64 + ldh + 8],  Kg + 8);
        cp_b128(&Kl[ldr * 64 + ldh + 16], Kg + 16);
        cp_b128(&Kl[ldr * 64 + ldh + 24], Kg + 24);
        // V tile: load to registers, transpose-scatter into LDS
        const u16* Vg = vb + ((size_t)bh * SEQ + kt + ldr) * HDIM + ldh;
        V32 vv;
        vv.q[0] = *(const uint4*)(Vg);
        vv.q[1] = *(const uint4*)(Vg + 8);
        vv.q[2] = *(const uint4*)(Vg + 16);
        vv.q[3] = *(const uint4*)(Vg + 24);
        for (int e = 0; e < 32; e++)
            Vt[(ldh + e) * 128 + ldr] = vv.h[e];
        cp_wait();
        __syncthreads();

        // S = Q K^T : 8 column subtiles x (K=64 -> 2 wmma steps)
        v8f s[8];
        for (int j = 0; j < 8; j++) { v8f z = {}; s[j] = z; }
        for (int j = 0; j < 8; j++) {
            Frag bk0, bk1;
            const u16* p0 = &Kl[(j * 16 + r) * 64 + 0 * 32 + g * 8];
            bk0.q[0] = *(const uint4*)p0;  bk0.q[1] = *(const uint4*)(p0 + 16);
            const u16* p1 = &Kl[(j * 16 + r) * 64 + 1 * 32 + g * 8];
            bk1.q[0] = *(const uint4*)p1;  bk1.q[1] = *(const uint4*)(p1 + 16);
            s[j] = wmma_bf16(aq[0], bk0, s[j]);
            s[j] = wmma_bf16(aq[1], bk1, s[j]);
        }

        // online softmax: whole 128-wide row lives in this wave (16-lane shuffles only)
        float alpha[8];
        for (int v = 0; v < 8; v++) {
            float t = s[0][v];
            for (int j = 1; j < 8; j++) t = fmaxf(t, s[j][v]);
            for (int off = 1; off < 16; off <<= 1) t = fmaxf(t, __shfl_xor(t, off, 32));
            float mnew = fmaxf(mi_[v], t * cscale);
            alpha[v] = exp2f(mi_[v] - mnew);
            float rs = 0.f;
            for (int j = 0; j < 8; j++) {
                float p0 = exp2f(s[j][v] * cscale - mnew);
                s[j][v] = p0;
                rs += p0;
            }
            for (int off = 1; off < 16; off <<= 1) rs += __shfl_xor(rs, off, 32);
            li_[v] = li_[v] * alpha[v] + rs;
            mi_[v] = mnew;
        }
        for (int dj = 0; dj < 4; dj++)
            for (int v = 0; v < 8; v++) o[dj][v] *= alpha[v];

        // transpose P (C-layout) -> row-major bf16 in per-wave LDS
        for (int j = 0; j < 8; j++)
            for (int v = 0; v < 8; v++)
                Pw[(v + g * 8) * 128 + j * 16 + r] = f2bf(s[j][v]);

        // O += P V : K=128 -> 4 wmma steps, 4 d-subtiles
        for (int kk = 0; kk < 4; kk++) {
            Frag ap;
            const u16* pp = &Pw[r * 128 + kk * 32 + g * 8];
            ap.q[0] = *(const uint4*)pp;
            ap.q[1] = *(const uint4*)(pp + 16);
            for (int dj = 0; dj < 4; dj++) {
                Frag bv;
                const u16* vp = &Vt[(dj * 16 + r) * 128 + kk * 32 + g * 8];
                bv.q[0] = *(const uint4*)vp;
                bv.q[1] = *(const uint4*)(vp + 16);
                o[dj] = wmma_bf16(ap, bv, o[dj]);
            }
        }
    }

    // epilogue: O / l, write bf16 into [B, N, DIM] (token-major, heads concatenated)
    const int b = bh / NHEADS, h = bh % NHEADS;
    for (int v = 0; v < 8; v++) {
        float inv = 1.f / li_[v];
        int row = qt * 128 + wave * 16 + v + g * 8;
        u16* dst = attnb + ((size_t)(b * SEQ + row)) * DIMC + h * HDIM;
        for (int dj = 0; dj < 4; dj++)
            dst[dj * 16 + r] = f2bf(o[dj][v] * inv);
    }
}

// ---------------- Host-side orchestration ----------------
extern "C" void kernel_launch(void* const* d_in, const int* in_sizes, int n_in,
                              void* d_out, int out_size, void* d_ws, size_t ws_size,
                              hipStream_t stream) {
    (void)in_sizes; (void)n_in; (void)out_size; (void)ws_size;
    const float* x      = (const float*)d_in[0];
    const float* qkv_w  = (const float*)d_in[1];
    const float* qkv_b  = (const float*)d_in[2];
    const float* proj_w = (const float*)d_in[3];
    const float* proj_b = (const float*)d_in[4];
    float* out = (float*)d_out;

    char* ws = (char*)d_ws;
    size_t off = 0;
    u16* xb     = (u16*)(ws + off); off += (size_t)MTOT * DIMC * 2;   // 25.2 MB
    u16* wqkvb  = (u16*)(ws + off); off += (size_t)QKVN * DIMC * 2;   //  3.5 MB
    u16* wprojb = (u16*)(ws + off); off += (size_t)DIMC * DIMC * 2;   //  1.2 MB
    u16* q_bh   = (u16*)(ws + off); off += (size_t)MTOT * DIMC * 2;   // 25.2 MB
    u16* k_bh   = (u16*)(ws + off); off += (size_t)MTOT * DIMC * 2;
    u16* v_bh   = (u16*)(ws + off); off += (size_t)MTOT * DIMC * 2;
    u16* attnb  = (u16*)(ws + off); off += (size_t)MTOT * DIMC * 2;

    // Pass 0: bf16 staging
    int nx = MTOT * DIMC / 4, nw = QKVN * DIMC / 4, np = DIMC * DIMC / 4;
    cvt_f32_bf16<<<(nx + 255) / 256, 256, 0, stream>>>(x, xb, nx);
    cvt_f32_bf16<<<(nw + 255) / 256, 256, 0, stream>>>(qkv_w, wqkvb, nw);
    cvt_f32_bf16<<<(np + 255) / 256, 256, 0, stream>>>(proj_w, wprojb, np);

    // Pass 1: QKV projection (M=16384, N=2304, K=768), scatter q/k/v
    gemm_bf16<<<dim3(QKVN / 128, MTOT / 128), 256, 0, stream>>>(
        xb, wqkvb, qkv_b, q_bh, k_bh, v_bh, nullptr, DIMC, 0);

    // Pass 2: attention (8 q-tiles x 192 (b,h) pairs)
    flash_attn<<<dim3(SEQ / 128, BATCH * NHEADS), 256, 0, stream>>>(
        q_bh, k_bh, v_bh, attnb);

    // Pass 3: output projection (M=16384, N=768, K=768), f32 out
    gemm_bf16<<<dim3(DIMC / 128, MTOT / 128), 256, 0, stream>>>(
        attnb, wprojb, proj_b, nullptr, nullptr, nullptr, out, DIMC, 1);
}
